// ConvBNN_33749853012664
// MI455X (gfx1250) — 8-bit matrix operand order repaired and checked by reading, compiled, not run on hardware
//
#include <hip/hip_runtime.h>


typedef __attribute__((ext_vector_type(16))) int   v16i;
typedef __attribute__((ext_vector_type(8)))  float v8f;

__device__ __forceinline__ unsigned char sgn8(float z) {
  return z > 0.f ? (unsigned char)0x38 : (z < 0.f ? (unsigned char)0xB8 : (unsigned char)0x00);
}

__global__ void bn_prep(const float* __restrict__ bn, float* __restrict__ out,
                        int C, int Cpad) {
  int c = blockIdx.x * blockDim.x + threadIdx.x;
  if (c >= Cpad) return;
  if (c < C) {
    float g = bn[c], b = bn[C + c], m = bn[2 * C + c], v = bn[3 * C + c];
    float inv = g * rsqrtf(v + 1e-5f);
    out[2 * c] = inv;
    out[2 * c + 1] = b - m * inv;
  } else {
    out[2 * c] = 0.f;
    out[2 * c + 1] = 0.f;
  }
}

__global__ void pack_conv_w(const float* __restrict__ w, unsigned char* __restrict__ out,
                            int O, int I) {
  int idx = blockIdx.x * blockDim.x + threadIdx.x;
  int total = O * I * 9;
  if (idx >= total) return;
  int o = idx / (I * 9);
  int r = idx - o * (I * 9);
  int i = r / 9;
  int s = r - i * 9;
  out[(size_t)o * (9 * I) + (size_t)s * I + i] = sgn8(w[idx]);
}

__global__ void pack_fc1(const float* __restrict__ w, unsigned char* __restrict__ out) {
  int idx = blockIdx.x * blockDim.x + threadIdx.x;
  if (idx >= 2048 * 4608) return;
  int o = idx / 4608;
  int r = idx - o * 4608;
  int c = r / 9;
  int s = r - c * 9;
  out[(size_t)o * 4608 + (size_t)s * 512 + c] = sgn8(w[idx]);
}

__global__ void pack_fc2(const float* __restrict__ w, unsigned char* __restrict__ out, int total) {
  int idx = blockIdx.x * blockDim.x + threadIdx.x;
  if (idx >= total) return;
  out[idx] = sgn8(w[idx]);
}

__global__ void pack_fc3(const float* __restrict__ w, unsigned char* __restrict__ out) {
  int idx = blockIdx.x * blockDim.x + threadIdx.x;
  if (idx >= 16 * 2048) return;
  int o = idx >> 11;
  int k = idx & 2047;
  out[idx] = (o < 10) ? sgn8(w[o * 2048 + k]) : (unsigned char)0;
}

__global__ __launch_bounds__(256) void conv1_kernel(
    const float* __restrict__ x, const float* __restrict__ w,
    const float* __restrict__ bnc, unsigned char* __restrict__ out) {
  int idx = blockIdx.x * 256 + threadIdx.x;
  int oc = idx & 127;
  int pix = idx >> 7;
  int n = pix >> 10;
  int rem = pix & 1023;
  int oy = rem >> 5, ox = rem & 31;
  float acc = 0.f;
#pragma unroll
  for (int ci = 0; ci < 3; ++ci) {
#pragma unroll
    for (int dy = 0; dy < 3; ++dy) {
      int iy = oy + dy - 1;
      if (iy < 0 || iy >= 32) continue;
#pragma unroll
      for (int dx = 0; dx < 3; ++dx) {
        int ix = ox + dx - 1;
        if (ix < 0 || ix >= 32) continue;
        float wv = w[((oc * 3 + ci) * 3 + dy) * 3 + dx];
        float bw = (wv > 0.f) ? 1.f : ((wv < 0.f) ? -1.f : 0.f);
        acc += bw * x[((n * 3 + ci) * 32 + iy) * 32 + ix];
      }
    }
  }
  float z = acc * bnc[2 * oc] + bnc[2 * oc + 1];
  out[idx] = sgn8(z);
}

template <int POOL>
__device__ __forceinline__ void row_to_pixel(int RA, int OH, int OW,
                                             int& n, int& oy, int& ox) {
  if (POOL) {
    const int PH = OH >> 1, PW = OW >> 1;
    int cell = RA >> 2, p = RA & 3;
    n = cell / (PH * PW);
    int r = cell - n * (PH * PW);
    int ph = r / PW, pw = r - ph * PW;
    oy = 2 * ph + (p >> 1);
    ox = 2 * pw + (p & 1);
  } else {
    n = RA / (OH * OW);
    int r = RA - n * (OH * OW);
    oy = r / OW;
    ox = r - oy * OW;
  }
}

template <int POOL, int PAD, int MT>
__global__ __launch_bounds__(256) void conv_wmma_kernel(
    const unsigned char* __restrict__ in, const unsigned char* __restrict__ wpk,
    const float* __restrict__ bnc, unsigned char* __restrict__ out,
    int N, int H, int W, int Cin, int Cout, int OH, int OW) {
  constexpr int NG = 8 / MT;
  constexpr int BM = MT * 32;
  __shared__ unsigned char sA[2][BM * 128];

  const int tid = threadIdx.x;
  const int lane = tid & 31;
  const int wv = tid >> 5;
  const int mt = wv / NG;
  const int ng = wv - mt * NG;
  const int row = lane & 15;
  const int hi = lane >> 4;
  const int Mbase = blockIdx.x * BM;
  const int K = 9 * Cin;
  const int chunks = K >> 7;

  int st_n[MT], st_oy[MT], st_ox[MT], st_k[MT], st_f[MT];
#pragma unroll
  for (int sgi = 0; sgi < MT; ++sgi) {
    int f = (tid + sgi * 256) * 16;
    st_f[sgi] = f;
    st_k[sgi] = f & 127;
    row_to_pixel<POOL>(Mbase + (f >> 7), OH, OW, st_n[sgi], st_oy[sgi], st_ox[sgi]);
  }

  auto stage = [&](int kc) {
    const int kbase = kc << 7;
    const int s = kbase / Cin;
    const int ci = kbase - s * Cin;
    const int dy = s / 3, dx = s - dy * 3;
    unsigned char* buf = sA[kc & 1];
#pragma unroll
    for (int sgi = 0; sgi < MT; ++sgi) {
      int iy = st_oy[sgi] + dy - PAD, ix = st_ox[sgi] + dx - PAD;
      if (iy >= 0 && iy < H && ix >= 0 && ix < W) {
        const unsigned char* gp =
            in + (size_t)(((st_n[sgi] * H + iy) * W) + ix) * Cin + ci + st_k[sgi];
        unsigned ldsoff = (unsigned)(unsigned long long)(&buf[st_f[sgi]]);
        asm volatile("global_load_async_to_lds_b128 %0, %1, off"
                     :: "v"(ldsoff), "v"(gp) : "memory");
      } else {
        int4 zz = {0, 0, 0, 0};
        *(int4*)(&buf[st_f[sgi]]) = zz;
      }
    }
  };

  v8f acc[2][4];
  const v8f z8 = {0.f, 0.f, 0.f, 0.f, 0.f, 0.f, 0.f, 0.f};
#pragma unroll
  for (int st = 0; st < 2; ++st)
#pragma unroll
    for (int nc = 0; nc < 4; ++nc) acc[st][nc] = z8;

  stage(0);
  for (int kc = 0; kc < chunks; ++kc) {
    const int kbase = kc << 7;
    asm volatile("s_wait_asynccnt 0x0" ::: "memory");
    __syncthreads();
    if (kc + 1 < chunks) stage(kc + 1);

    const unsigned char* bufr = sA[kc & 1];
    v16i a[2];
#pragma unroll
    for (int st = 0; st < 2; ++st) {
      const unsigned char* lp = &bufr[(mt * 32 + st * 16 + row) * 128 + (hi << 4)];
#pragma unroll
      for (int j = 0; j < 8; ++j) {
        unsigned long long dq = *(const unsigned long long*)(lp + (((j >> 1) << 5) + ((j & 1) << 3)));
        a[st][2 * j] = (int)(unsigned int)dq;
        a[st][2 * j + 1] = (int)(unsigned int)(dq >> 32);
      }
    }

#pragma unroll
    for (int nc = 0; nc < 4; ++nc) {
      const unsigned char* cbp =
          wpk + (size_t)(((ng * 4 + nc) * 16) + row) * K + kbase + (hi << 4);
      v16i b;
#pragma unroll
      for (int g = 0; g < 4; ++g) {
        int4 d = *(const int4*)(cbp + (g << 5));
        b[4 * g] = d.x; b[4 * g + 1] = d.y; b[4 * g + 2] = d.z; b[4 * g + 3] = d.w;
      }
      acc[0][nc] = __builtin_amdgcn_wmma_f32_16x16x128_fp8_fp8(a[0], b, (short)0, acc[0][nc], false, false);
      acc[1][nc] = __builtin_amdgcn_wmma_f32_16x16x128_fp8_fp8(a[1], b, (short)0, acc[1][nc], false, false);
    }
  }

#pragma unroll
  for (int st = 0; st < 2; ++st) {
    const int R = Mbase + mt * 32 + st * 16;
#pragma unroll
    for (int nc = 0; nc < 4; ++nc) {
      const int oc = (ng * 4 + nc) * 16 + row;
      const float inv = bnc[2 * oc], bi = bnc[2 * oc + 1];
      if (POOL) {
        float r0 = fmaxf(fmaxf(acc[st][nc][0], acc[st][nc][1]),
                         fmaxf(acc[st][nc][2], acc[st][nc][3]));
        float r1 = fmaxf(fmaxf(acc[st][nc][4], acc[st][nc][5]),
                         fmaxf(acc[st][nc][6], acc[st][nc][7]));
        int cb0 = (R >> 2) + hi * 2;
        out[(size_t)cb0 * Cout + oc] = sgn8(r0 * inv + bi);
        out[(size_t)(cb0 + 1) * Cout + oc] = sgn8(r1 * inv + bi);
      } else {
#pragma unroll
        for (int r = 0; r < 8; ++r) {
          int pix = R + r + 8 * hi;
          out[(size_t)pix * Cout + oc] = sgn8(acc[st][nc][r] * inv + bi);
        }
      }
    }
  }
}

template <int FINAL, int NCOLS>
__global__ __launch_bounds__(256) void fc_wmma_kernel(
    const unsigned char* __restrict__ A, const unsigned char* __restrict__ Bm,
    const float* __restrict__ bnc, unsigned char* __restrict__ out8,
    float* __restrict__ outf, int K, int N, int Nvalid) {
  const int lane = threadIdx.x & 31;
  const int wave = (blockIdx.x * blockDim.x + threadIdx.x) >> 5;
  const int colgroups = (N >> 4) / NCOLS;
  const int tileM = wave / colgroups;
  const int ng = wave - tileM * colgroups;
  const int row = lane & 15;
  const int hi = lane >> 4;

  const unsigned char* rp = A + (size_t)(tileM * 16 + row) * K + (hi << 4);
  const int chunks = K >> 7;

  v8f acc[NCOLS];
  const v8f z8 = {0.f, 0.f, 0.f, 0.f, 0.f, 0.f, 0.f, 0.f};
#pragma unroll
  for (int nc = 0; nc < NCOLS; ++nc) acc[nc] = z8;

  for (int kc = 0; kc < chunks; ++kc) {
    const int kbase = kc << 7;
    v16i a;
#pragma unroll
    for (int j = 0; j < 8; ++j) {
      unsigned long long d = *(const unsigned long long*)(rp + kbase + (((j >> 1) << 5) + ((j & 1) << 3)));
      a[2 * j] = (int)(unsigned int)d;
      a[2 * j + 1] = (int)(unsigned int)(d >> 32);
    }
#pragma unroll
    for (int nc = 0; nc < NCOLS; ++nc) {
      const unsigned char* cb =
          Bm + (size_t)((ng * NCOLS + nc) * 16 + row) * K + kbase + (hi << 4);
      v16i b;
#pragma unroll
      for (int g = 0; g < 4; ++g) {
        int4 d = *(const int4*)(cb + (g << 5));
        b[4 * g] = d.x; b[4 * g + 1] = d.y; b[4 * g + 2] = d.z; b[4 * g + 3] = d.w;
      }
      acc[nc] = __builtin_amdgcn_wmma_f32_16x16x128_fp8_fp8(a, b, (short)0, acc[nc], false, false);
    }
  }

#pragma unroll
  for (int nc = 0; nc < NCOLS; ++nc) {
    const int nidx = (ng * NCOLS + nc) * 16 + row;
    const float inv = bnc[2 * nidx], bi = bnc[2 * nidx + 1];
#pragma unroll
    for (int r = 0; r < 8; ++r) {
      int m = tileM * 16 + r + 8 * hi;
      float z = acc[nc][r] * inv + bi;
      if (FINAL) {
        if (nidx < Nvalid) outf[m * Nvalid + nidx] = z;
      } else {
        out8[(size_t)m * N + nidx] = sgn8(z);
      }
    }
  }
}

extern "C" void kernel_launch(void* const* d_in, const int* in_sizes, int n_in,
                              void* d_out, int out_size, void* d_ws, size_t ws_size,
                              hipStream_t stream) {
  (void)in_sizes; (void)n_in; (void)out_size; (void)ws_size;
  const float* x = (const float*)d_in[0];
  const float* w1 = (const float*)d_in[1];
  const float* w2 = (const float*)d_in[2];
  const float* w3 = (const float*)d_in[3];
  const float* w4 = (const float*)d_in[4];
  const float* w5 = (const float*)d_in[5];
  const float* w6 = (const float*)d_in[6];
  const float* fw1 = (const float*)d_in[7];
  const float* fw2 = (const float*)d_in[8];
  const float* fw3 = (const float*)d_in[9];
  const float* bna[9];
  for (int i = 0; i < 9; ++i) bna[i] = (const float*)d_in[10 + i];

  unsigned char* base = (unsigned char*)d_ws;
  size_t off = 0;
  auto take = [&](size_t bytes) -> unsigned char* {
    unsigned char* p = base + off;
    off = (off + bytes + 255) & ~(size_t)255;
    return p;
  };
  unsigned char* act0 = take((size_t)256 * 32 * 32 * 128);
  unsigned char* act1 = take((size_t)256 * 32 * 32 * 128);
  unsigned char* pw2 = take(128 * 1152);
  unsigned char* pw3 = take(256 * 1152);
  unsigned char* pw4 = take(256 * 2304);
  unsigned char* pw5 = take(512 * 2304);
  unsigned char* pw6 = take((size_t)512 * 4608);
  unsigned char* pf1 = take((size_t)2048 * 4608);
  unsigned char* pf2 = take((size_t)2048 * 2048);
  unsigned char* pf3 = take(16 * 2048);
  float* bnc[9];
  const int bnC[9] = {128, 128, 256, 256, 512, 512, 2048, 2048, 10};
  const int bnP[9] = {128, 128, 256, 256, 512, 512, 2048, 2048, 16};
  for (int i = 0; i < 9; ++i) bnc[i] = (float*)take((size_t)bnP[i] * 2 * 4);

  dim3 blk(256);
  auto cdiv = [](int a, int b) { return (a + b - 1) / b; };

  pack_conv_w<<<cdiv(128 * 128 * 9, 256), blk, 0, stream>>>(w2, pw2, 128, 128);
  pack_conv_w<<<cdiv(256 * 128 * 9, 256), blk, 0, stream>>>(w3, pw3, 256, 128);
  pack_conv_w<<<cdiv(256 * 256 * 9, 256), blk, 0, stream>>>(w4, pw4, 256, 256);
  pack_conv_w<<<cdiv(512 * 256 * 9, 256), blk, 0, stream>>>(w5, pw5, 512, 256);
  pack_conv_w<<<cdiv(512 * 512 * 9, 256), blk, 0, stream>>>(w6, pw6, 512, 512);
  pack_fc1<<<cdiv(2048 * 4608, 256), blk, 0, stream>>>(fw1, pf1);
  pack_fc2<<<cdiv(2048 * 2048, 256), blk, 0, stream>>>(fw2, pf2, 2048 * 2048);
  pack_fc3<<<cdiv(16 * 2048, 256), blk, 0, stream>>>(fw3, pf3);
  for (int i = 0; i < 9; ++i)
    bn_prep<<<cdiv(bnP[i], 256), blk, 0, stream>>>(bna[i], bnc[i], bnC[i], bnP[i]);

  conv1_kernel<<<(256 * 1024 * 128) / 256, blk, 0, stream>>>(x, w1, bnc[0], act0);

  conv_wmma_kernel<1, 1, 4><<<2048, blk, 0, stream>>>(act0, pw2, bnc[1], act1,
                                                      256, 32, 32, 128, 128, 32, 32);
  conv_wmma_kernel<0, 1, 2><<<1024, blk, 0, stream>>>(act1, pw3, bnc[2], act0,
                                                      256, 16, 16, 128, 256, 16, 16);
  conv_wmma_kernel<1, 1, 2><<<1024, blk, 0, stream>>>(act0, pw4, bnc[3], act1,
                                                      256, 16, 16, 256, 256, 16, 16);
  conv_wmma_kernel<0, 1, 1><<<512, blk, 0, stream>>>(act1, pw5, bnc[4], act0,
                                                     256, 8, 8, 256, 512, 8, 8);
  conv_wmma_kernel<1, 0, 1><<<288, blk, 0, stream>>>(act0, pw6, bnc[5], act1,
                                                     256, 8, 8, 512, 512, 6, 6);

  fc_wmma_kernel<0, 4><<<64, blk, 0, stream>>>(act1, pf1, bnc[6], act0, nullptr,
                                               4608, 2048, 2048);
  fc_wmma_kernel<0, 4><<<64, blk, 0, stream>>>(act0, pf2, bnc[7], act1, nullptr,
                                               2048, 2048, 2048);
  fc_wmma_kernel<1, 1><<<2, blk, 0, stream>>>(act1, pf3, bnc[8], nullptr,
                                              (float*)d_out, 2048, 16, 10);
}
